// GroupedLinear_40922448396693
// MI455X (gfx1250) — matrix operand order repaired and checked by reading, not compiled, not run on hardware
//
#include <hip/hip_runtime.h>

typedef __bf16 bf16_t;
typedef __attribute__((ext_vector_type(16))) __bf16 v16bf;
typedef __attribute__((ext_vector_type(4)))  __bf16 bf16x4;
typedef __attribute__((ext_vector_type(8)))  __bf16 bf16x8;
typedef __attribute__((ext_vector_type(8)))  float   v8f;

// Dims fixed by the reference problem.
constexpr int Gn = 16;
constexpr int Kd = 2048;
constexpr int Nd = 2048;
constexpr int Td = 16384;

constexpr int BM = 128, BN = 256, BK = 32;
// LDS row stride in bf16 elems: 40 elems = 80 bytes -> 16B-aligned rows for
// b128 frag loads; bank stride 20 (gcd(20,64)=4) -> 16 distinct banks across
// the 16 rows touched by a quarter-wave.
constexpr int LDA = 40;
constexpr int LDB = 40;

__device__ __forceinline__ bf16x4 cvt4(float4 v) {
  bf16x4 r;
  r[0] = (__bf16)v.x;
  r[1] = (__bf16)v.y;
  r[2] = (__bf16)v.z;
  r[3] = (__bf16)v.w;
  return r;
}

__global__ __launch_bounds__(256, 1)
void grouped_gemm_bf16_wmma(const float* __restrict__ hs,
                            const float* __restrict__ wgt,
                            const int* __restrict__ offsets,
                            const int* __restrict__ tpe,
                            float* __restrict__ out) {
  __shared__ bf16_t lA[2][BM * LDA];  // A tile, K-major rows (bf16)
  __shared__ bf16_t lB[2][BN * LDB];  // B tile, N-major rows (bf16)

  const int g     = blockIdx.z;
  const int rows  = tpe[g];
  const int row0l = blockIdx.x * BM;
  if (row0l >= rows) return;  // uniform per block: safe before barriers
  const int row0 = offsets[g] + row0l;
  const int col0 = blockIdx.y * BN;

  const int tid  = threadIdx.x;
  const int lane = tid & 31;
  const int wave = tid >> 5;
  const int wrow = (wave & 1) * 64;   // 2 row-waves x 64 rows
  const int wcol = (wave >> 1) * 64;  // 4 col-waves x 64 cols
  const int l15  = lane & 15;
  const int lhi  = lane >> 4;

  const float* wg = wgt + (size_t)g * Kd * Nd;

  // Cooperative-load coordinates (constant over the K loop).
  const int aR  = tid >> 3;        // + i*32 ; 8 float4 per 32-float A row
  const int aC  = (tid & 7) * 4;   // float col within BK
  const int bN0 = (tid & 127) * 2; // 2 n-columns per thread
  const int bKh = (tid >> 7) * 16; // 16 k-values per column

  float4 aReg[4];
  float2 bReg[16];

  auto loadGlobal = [&](int kk) {
#pragma unroll
    for (int i = 0; i < 4; ++i) {
      const int r = aR + i * 32;
      if (row0l + r < rows)
        aReg[i] = *(const float4*)(hs + (size_t)(row0 + r) * Kd + kk + aC);
      else
        aReg[i] = make_float4(0.f, 0.f, 0.f, 0.f);
    }
    // B: per k-row, lanes cover 64 consecutive floats (256B) -> coalesced.
#pragma unroll
    for (int j = 0; j < 16; ++j)
      bReg[j] = *(const float2*)(wg + (size_t)(kk + bKh + j) * Nd + col0 + bN0);
  };

  auto storeLds = [&](int buf) {
#pragma unroll
    for (int i = 0; i < 4; ++i) {
      const int r = aR + i * 32;
      *(bf16x4*)&lA[buf][r * LDA + aC] = cvt4(aReg[i]);  // b64 store
    }
    // Transpose-in-registers: per n-column, 16 k-values contiguous in LDS.
#pragma unroll
    for (int s = 0; s < 2; ++s) {
#pragma unroll
      for (int h = 0; h < 2; ++h) {
        bf16x8 p;
#pragma unroll
        for (int j = 0; j < 8; ++j) {
          const float2 v = bReg[h * 8 + j];
          p[j] = (__bf16)(s == 0 ? v.x : v.y);
        }
        *(bf16x8*)&lB[buf][(bN0 + s) * LDB + bKh + h * 8] = p;  // b128 store
      }
    }
  };

  v8f acc[4][4] = {};
  union Frag { v16bf v; int4 q[2]; };

  auto compute = [&](int buf) {
    Frag a[4], b[4];
    // A 16x32 bf16 frag: lane holds row m, K ks..ks+15 contiguous (ks = lhi*16)
#pragma unroll
    for (int mi = 0; mi < 4; ++mi) {
      const bf16_t* p = &lA[buf][(wrow + mi * 16 + l15) * LDA + lhi * 16];
      a[mi].q[0] = *(const int4*)p;
      a[mi].q[1] = *(const int4*)(p + 8);
    }
    // B 32x16 bf16 frag: lane holds column n, K ks..ks+15 contiguous (ks = lhi*16)
#pragma unroll
    for (int ni = 0; ni < 4; ++ni) {
      const bf16_t* p = &lB[buf][(wcol + ni * 16 + l15) * LDB + lhi * 16];
      b[ni].q[0] = *(const int4*)p;
      b[ni].q[1] = *(const int4*)(p + 8);
    }
#pragma unroll
    for (int mi = 0; mi < 4; ++mi)
#pragma unroll
      for (int ni = 0; ni < 4; ++ni)
        acc[mi][ni] = __builtin_amdgcn_wmma_f32_16x16x32_bf16(
            false, a[mi].v, false, b[ni].v, (short)0, acc[mi][ni], false, false);
  };

  constexpr int NIT = Kd / BK;  // 64
  loadGlobal(0);
  storeLds(0);
  __syncthreads();
#pragma unroll 1
  for (int it = 0; it < NIT; ++it) {
    if (it + 1 < NIT) loadGlobal((it + 1) * BK);  // keep HBM loads in flight
    compute(it & 1);
    if (it + 1 < NIT) {
      storeLds((it + 1) & 1);
      __syncthreads();
    }
  }

  // Epilogue: D layout — VGPR j: lanes0-15 M=j, lanes16-31 M=8+j; N = lane&15.
#pragma unroll
  for (int mi = 0; mi < 4; ++mi) {
#pragma unroll
    for (int ni = 0; ni < 4; ++ni) {
      const int n = col0 + wcol + ni * 16 + l15;
#pragma unroll
      for (int j = 0; j < 8; ++j) {
        const int ml = row0l + wrow + mi * 16 + lhi * 8 + j;
        if (ml < rows)
          out[(size_t)(offsets[g] + ml) * Nd + n] = acc[mi][ni][j];
      }
    }
  }
}

extern "C" void kernel_launch(void* const* d_in, const int* in_sizes, int n_in,
                              void* d_out, int out_size, void* d_ws, size_t ws_size,
                              hipStream_t stream) {
  (void)in_sizes; (void)n_in; (void)d_ws; (void)ws_size; (void)out_size;
  const float* hs      = (const float*)d_in[0];
  const float* w       = (const float*)d_in[1];
  const int*   offsets = (const int*)d_in[2];
  // d_in[3] = blockscale_offsets (unused by reference math path)
  const int*   tpe     = (const int*)d_in[4];
  float* out = (float*)d_out;

  // grid.x over-provisions row tiles per group (worst case one group owns all
  // tokens); blocks past a group's token count early-exit on one scalar load.
  dim3 grid((Td + BM - 1) / BM, Nd / BN, Gn);
  dim3 block(256, 1, 1);
  hipLaunchKernelGGL(grouped_gemm_bf16_wmma, grid, block, 0, stream,
                     hs, w, offsets, tpe, out);
}
